// EGNNDynamics_83416854823129
// MI455X (gfx1250) — matrix operand order repaired and checked by reading, not compiled, not run on hardware
//
#include <hip/hip_runtime.h>
#include <math.h>

typedef _Float16 h16;
typedef __attribute__((ext_vector_type(16))) _Float16 v16h;
typedef __attribute__((ext_vector_type(8)))  float    v8f;

#define NNODE 2048
#define FEAT  64
#define EPN   127
#define KIN   192
#define LDI   200   // padded leading dim for 192-wide tiles (halves); 400B rows, 16B aligned
#define LDM   72    // padded leading dim for 64-wide tiles (halves); 144B rows, 16B aligned
#define LDC   136   // padded leading dim for 128-wide tiles (halves); 272B rows, 16B aligned

// fast transcendentals: v_rcp_f32 / v_exp_f32 are TRANS ops that co-execute with WMMA
__device__ __forceinline__ float sigm_(float x){ return __builtin_amdgcn_rcpf(1.0f + __expf(-x)); }
__device__ __forceinline__ float silu_(float x){ return x * sigm_(x); }
__device__ __forceinline__ float tanh_(float x){
  float e = __expf(-2.0f*x);
  return (1.0f - e) * __builtin_amdgcn_rcpf(1.0f + e);
}

// ---- WMMA fragment loaders per CDNA5 ISA 7.12.2 (wave32) ----
__device__ __forceinline__ v16h frag_a(const h16* __restrict__ tile, int ld, int k0, int lane){
  const int m  = lane & 15;
  const int kh = (lane >> 4) << 3;
  const h16* rp = tile + m*ld + k0;
  v16h a;
#pragma unroll
  for (int v=0; v<8; ++v){
    const int kk = (v < 4) ? (kh + 2*v) : (16 + kh + 2*(v-4));
    a[2*v]   = rp[kk];
    a[2*v+1] = rp[kk+1];
  }
  return a;
}
__device__ __forceinline__ v16h frag_b(const h16* __restrict__ w, int ld, int k0, int lane){
  const int n  = lane & 15;
  const int kh = (lane >> 4) << 3;
  const h16* rp = w + n*ld + k0 + kh;
  v16h b;
#pragma unroll
  for (int j=0;j<16;++j) b[j] = rp[(j < 8) ? j : (8 + j)];
  return b;
}

// =====================================================================
// Kernel 0: f32 -> f16 weight conversion (once per launch)
// =====================================================================
__global__ __launch_bounds__(256) void egnn_cvt16(
    const float* __restrict__ src, h16* __restrict__ dst, int n)
{
  int i = blockIdx.x*blockDim.x + threadIdx.x;
  if (i < n) dst[i] = (h16)src[i];
}

// =====================================================================
// Kernel 1: node embedding  h = [one_hot(type) | gauss(t)] @ hW^T + hb
// =====================================================================
__global__ __launch_bounds__(256) void egnn_embed(
    const float* __restrict__ t, const int* __restrict__ ptypes,
    const float* __restrict__ tmean, const float* __restrict__ tstd,
    const float* __restrict__ hW, const float* __restrict__ hb,
    float* __restrict__ h, h16* __restrict__ h16out)
{
  int idx = blockIdx.x*blockDim.x + threadIdx.x;
  if (idx >= NNODE*FEAT) return;
  int n = idx >> 6, f = idx & 63;
  int ty = ptypes[n & 127];
  float tv = t[0];
  float acc = hb[f] + hW[f*21 + ty];
#pragma unroll
  for (int k=0;k<16;++k){
    float z = (tv - tmean[k]) / tstd[k];
    acc += __expf(-0.5f*z*z) * hW[f*21 + 5 + k];
  }
  h[idx] = acc;
  h16out[idx] = (h16)acc;
}

// =====================================================================
// Kernel 2: per-destination-node fused edge pipeline (one block per dst)
// =====================================================================
__global__ __launch_bounds__(256) void egnn_edge_wmma(
    const h16*   __restrict__ hfe,   const float* __restrict__ pos,
    const int*   __restrict__ esrc,
    const float* __restrict__ dmean, const float* __restrict__ dstd,
    const h16*   __restrict__ W1h,   const float* __restrict__ mb1,
    const h16*   __restrict__ W2h,   const float* __restrict__ mb2,
    const float* __restrict__ aW,    const float* __restrict__ ab,
    const h16*   __restrict__ X1h,   const float* __restrict__ xb1,
    const float* __restrict__ xW2,
    h16* __restrict__ agg16, float* __restrict__ posOut)
{
  extern __shared__ unsigned char smemraw[];
  h16*   sInp  = (h16*)smemraw;            // 128 x LDI
  h16*   sW1   = sInp  + 128*LDI;          // 64  x LDI   (mW1)
  h16*   sBufA = sW1   + 64*LDI;           // 128 x LDM
  h16*   sBufB = sBufA + 128*LDM;          // 128 x LDM
  h16*   sW2   = sBufB + 128*LDM;          // 64  x LDM   (mW2, then xW1)
  float* sDM   = (float*)(sW2 + 64*LDM);   // 128 x 4 : dir xyz + mask
  float* sDisp = sDM + 128*4;              // 128 x 4 : per-edge disp
  float* sRed  = sDisp + 128*4;            // 256     : agg partials

  const int dst  = blockIdx.x;
  const int tid  = threadIdx.x;
  const int lane = tid & 31;
  const int wave = tid >> 5;

  __builtin_prefetch(W1h + tid*96, 0, 3);   // global_prefetch_b8: warm weights

  // stage pre-converted f16 weights via aligned 16B vector copies
  {
    const uint4* g1 = (const uint4*)W1h;   // 64 rows x 24 chunks
    for (int i = tid; i < 64*24; i += 256){
      int r = i / 24, c = i % 24;
      *(uint4*)(sW1 + r*LDI + c*8) = g1[i];
    }
    const uint4* g2 = (const uint4*)W2h;   // 64 rows x 8 chunks
    for (int i = tid; i < 64*8; i += 256){
      int r = i >> 3, c = i & 7;
      *(uint4*)(sW2 + r*LDM + c*8) = g2[i];
    }
  }

  // edge setup: 2 threads per row, 128 rows (row 127 = zero pad)
  {
    const int row  = tid >> 1;
    const int part = tid & 1;
    if (row < EPN) {
      const int e   = dst*EPN + row;
      const int src = esrc[e];
      const float dx = pos[dst*3+0]-pos[src*3+0];
      const float dy = pos[dst*3+1]-pos[src*3+1];
      const float dz = pos[dst*3+2]-pos[src*3+2];
      const float d2 = dx*dx+dy*dy+dz*dz;
      const float dd = sqrtf(d2);
      const float inv = (d2 > 0.f) ? __builtin_amdgcn_rcpf(dd) : 0.f;
      if (part == 0) {
        const uint4* hd = (const uint4*)(hfe + dst*64);
        const uint4* hs = (const uint4*)(hfe + src*64);
#pragma unroll
        for (int c=0;c<8;++c) *(uint4*)(sInp + row*LDI + c*8)      = hd[c];
#pragma unroll
        for (int c=0;c<8;++c) *(uint4*)(sInp + row*LDI + 64 + c*8) = hs[c];
        sDM[row*4+0]=dx*inv; sDM[row*4+1]=dy*inv; sDM[row*4+2]=dz*inv;
        sDM[row*4+3]=(dd<=2.0f)?1.f:0.f;
      } else {
        const float dc  = fminf(dd, 2.0f);
        const float cut = 0.5f*(__cosf(1.57079632679489662f*dc)+1.0f);
        for (int g=0; g<64; ++g){
          float z = (dd - dmean[g]) / dstd[g];
          sInp[row*LDI + 128 + g] = (h16)(__expf(-0.5f*z*z)*cut);
        }
      }
    } else { // pad row 127
      const uint4 z4 = {0,0,0,0};
      if (part == 0){
#pragma unroll
        for (int c=0;c<16;++c) *(uint4*)(sInp + row*LDI + c*8) = z4;
        sDM[row*4+0]=0.f; sDM[row*4+1]=0.f; sDM[row*4+2]=0.f; sDM[row*4+3]=0.f;
      } else {
#pragma unroll
        for (int c=16;c<24;++c) *(uint4*)(sInp + row*LDI + c*8) = z4;
      }
    }
  }
  __syncthreads();

  // GEMM1: hidden(128x64) = inp(128x192) @ mW1^T ; +mb1 ; SiLU -> sBufA
  {
    v16h af[6];
#pragma unroll
    for (int kc=0;kc<6;++kc) af[kc] = frag_a(sInp + wave*16*LDI, LDI, kc*32, lane);
#pragma unroll
    for (int nt = 0; nt < 4; ++nt){
      v8f acc = {};
#pragma unroll
      for (int kc = 0; kc < 6; ++kc){
        v16h b = frag_b(sW1 + nt*16*LDI, LDI, kc*32, lane);
        acc = __builtin_amdgcn_wmma_f32_16x16x32_f16(false,af[kc],false,b,(short)0,acc,false,false);
      }
      const int   n    = nt*16 + (lane & 15);
      const float bias = mb1[n];
      const int   mr   = wave*16 + ((lane>>4)<<3);
#pragma unroll
      for (int r=0;r<8;++r) sBufA[(mr+r)*LDM + n] = (h16)silu_(acc[r]+bias);
    }
  }
  __syncthreads();

  // GEMM2: m(128x64) = hidden @ mW2^T ; +mb2 ; SiLU -> sBufB
  {
    v16h af[2];
#pragma unroll
    for (int kc=0;kc<2;++kc) af[kc] = frag_a(sBufA + wave*16*LDM, LDM, kc*32, lane);
#pragma unroll
    for (int nt = 0; nt < 4; ++nt){
      v8f acc = {};
#pragma unroll
      for (int kc = 0; kc < 2; ++kc){
        v16h b = frag_b(sW2 + nt*16*LDM, LDM, kc*32, lane);
        acc = __builtin_amdgcn_wmma_f32_16x16x32_f16(false,af[kc],false,b,(short)0,acc,false,false);
      }
      const int   n    = nt*16 + (lane & 15);
      const float bias = mb2[n];
      const int   mr   = wave*16 + ((lane>>4)<<3);
#pragma unroll
      for (int r=0;r<8;++r) sBufB[(mr+r)*LDM + n] = (h16)silu_(acc[r]+bias);
    }
  }
  __syncthreads();

  // attention gate per edge (threads 0..127) ; threads 128..255 stage xW1
  if (tid < 128) {
    const int row = tid;
    float dot = ab[0];
    for (int c=0;c<64;++c) dot += (float)sBufB[row*LDM+c] * aW[c];
    const float g = sigm_(dot);
    for (int c=0;c<64;++c) sBufB[row*LDM+c] = (h16)((float)sBufB[row*LDM+c]*g);
  } else {
    const uint4* g3 = (const uint4*)X1h;
    for (int i = tid-128; i < 64*8; i += 128){
      int r = i >> 3, c = i & 7;
      *(uint4*)(sW2 + r*LDM + c*8) = g3[i];
    }
  }
  __syncthreads();

  // agg_h: masked column sums over edges (all 256 threads: 4 partials/col)
  {
    const int col = tid & 63, q = tid >> 6;
    float s = 0.f;
    for (int r = q*32; r < q*32+32; ++r) s += (float)sBufB[r*LDM+col] * sDM[r*4+3];
    sRed[tid] = s;
  }
  __syncthreads();
  if (tid < 64)
    agg16[dst*64+tid] = (h16)(sRed[tid] + sRed[tid+64] + sRed[tid+128] + sRed[tid+192]);

  // GEMM3: y(128x64) = m @ xW1^T ; +xb1 ; SiLU -> sBufA  (reads only sBufB/sW2)
  {
    v16h af[2];
#pragma unroll
    for (int kc=0;kc<2;++kc) af[kc] = frag_a(sBufB + wave*16*LDM, LDM, kc*32, lane);
#pragma unroll
    for (int nt = 0; nt < 4; ++nt){
      v8f acc = {};
#pragma unroll
      for (int kc = 0; kc < 2; ++kc){
        v16h b = frag_b(sW2 + nt*16*LDM, LDM, kc*32, lane);
        acc = __builtin_amdgcn_wmma_f32_16x16x32_f16(false,af[kc],false,b,(short)0,acc,false,false);
      }
      const int   n    = nt*16 + (lane & 15);
      const float bias = xb1[n];
      const int   mr   = wave*16 + ((lane>>4)<<3);
#pragma unroll
      for (int r=0;r<8;++r) sBufA[(mr+r)*LDM + n] = (h16)silu_(acc[r]+bias);
    }
  }
  __syncthreads();

  // per-edge displacement scalar: tanh(y @ xW2^T) * mask * dirs
  if (tid < 128) {
    const int row = tid;
    float s = 0.f;
    for (int c=0;c<64;++c) s += (float)sBufA[row*LDM+c] * xW2[c];
    const float tv = tanh_(s) * sDM[row*4+3];
    sDisp[row*4+0] = sDM[row*4+0]*tv;
    sDisp[row*4+1] = sDM[row*4+1]*tv;
    sDisp[row*4+2] = sDM[row*4+2]*tv;
  }
  __syncthreads();

  if (tid < 3) {
    float s = 0.f;
    for (int r=0;r<EPN;++r) s += sDisp[r*4+tid];
    posOut[dst*3+tid] = pos[dst*3+tid] + s;
  }
}

// =====================================================================
// Kernel 3: node update  h' = h + MLP([h | agg_h])   (64 nodes / block)
// =====================================================================
__global__ __launch_bounds__(128) void egnn_node_wmma(
    const float* __restrict__ h,   const h16* __restrict__ hfe,
    const h16*   __restrict__ agg16,
    const h16*   __restrict__ U1h, const float* __restrict__ ub1,
    const h16*   __restrict__ U2h, const float* __restrict__ ub2,
    float* __restrict__ hOut, h16* __restrict__ hOut16)
{
  __shared__ __align__(16) h16 sCat[64*LDC];
  __shared__ __align__(16) h16 sU1 [64*LDC];
  __shared__ __align__(16) h16 sT  [64*LDM];
  __shared__ __align__(16) h16 sU2 [64*LDM];
  const int tid=threadIdx.x, lane=tid&31, wave=tid>>5;
  const int base = blockIdx.x*64;

  for (int i=tid;i<64*8;i+=128){ int n=i>>3,c=i&7;
    *(uint4*)(sCat + n*LDC + c*8)      = ((const uint4*)(hfe   + (base+n)*64))[c];
    *(uint4*)(sCat + n*LDC + 64 + c*8) = ((const uint4*)(agg16 + (base+n)*64))[c];
  }
  { const uint4* g1 = (const uint4*)U1h;   // 64 rows x 16 chunks
    for (int i=tid;i<64*16;i+=128){ int n=i>>4,c=i&15; *(uint4*)(sU1 + n*LDC + c*8)=g1[i]; }
    const uint4* g2 = (const uint4*)U2h;   // 64 rows x 8 chunks
    for (int i=tid;i<64*8; i+=128){ int n=i>>3,c=i&7;  *(uint4*)(sU2 + n*LDM + c*8)=g2[i]; } }
  __syncthreads();

  {
    v16h af[4];
#pragma unroll
    for (int kc=0;kc<4;++kc) af[kc] = frag_a(sCat + wave*16*LDC, LDC, kc*32, lane);
#pragma unroll
    for (int nt=0;nt<4;++nt){
      v8f acc = {};
#pragma unroll
      for (int kc=0;kc<4;++kc){
        v16h b = frag_b(sU1 + nt*16*LDC, LDC, kc*32, lane);
        acc = __builtin_amdgcn_wmma_f32_16x16x32_f16(false,af[kc],false,b,(short)0,acc,false,false);
      }
      const int   n    = nt*16 + (lane & 15);
      const float bias = ub1[n];
      const int   mr   = wave*16 + ((lane>>4)<<3);
#pragma unroll
      for (int r=0;r<8;++r) sT[(mr+r)*LDM + n] = (h16)silu_(acc[r]+bias);
    }
  }
  __syncthreads();

  {
    v16h af[2];
#pragma unroll
    for (int kc=0;kc<2;++kc) af[kc] = frag_a(sT + wave*16*LDM, LDM, kc*32, lane);
#pragma unroll
    for (int nt=0;nt<4;++nt){
      v8f acc = {};
#pragma unroll
      for (int kc=0;kc<2;++kc){
        v16h b = frag_b(sU2 + nt*16*LDM, LDM, kc*32, lane);
        acc = __builtin_amdgcn_wmma_f32_16x16x32_f16(false,af[kc],false,b,(short)0,acc,false,false);
      }
      const int   n    = nt*16 + (lane & 15);
      const float bias = ub2[n];
      const int   mr   = wave*16 + ((lane>>4)<<3);
#pragma unroll
      for (int r=0;r<8;++r){
        const int node = base + mr + r;
        const float v = h[node*64+n] + acc[r] + bias;
        hOut[node*64+n]   = v;
        hOut16[node*64+n] = (h16)v;
      }
    }
  }
}

// =====================================================================
// Kernel 4: velocity = pos_final - pos0
// =====================================================================
__global__ __launch_bounds__(256) void egnn_vel(
    const float* __restrict__ posF, const float* __restrict__ x, float* __restrict__ out)
{
  int i = blockIdx.x*blockDim.x + threadIdx.x;
  if (i < NNODE*3) out[i] = posF[i] - x[i];
}

extern "C" void kernel_launch(void* const* d_in, const int* in_sizes, int n_in,
                              void* d_out, int out_size, void* d_ws, size_t ws_size,
                              hipStream_t stream) {
  (void)in_sizes; (void)n_in; (void)out_size; (void)ws_size;
  const float* t      = (const float*)d_in[0];
  const float* x      = (const float*)d_in[1];
  const int*   ptypes = (const int*)  d_in[2];
  const int*   esrc   = (const int*)  d_in[4];
  const float* tmean  = (const float*)d_in[5];
  const float* tstd   = (const float*)d_in[6];
  const float* hW     = (const float*)d_in[7];
  const float* hb     = (const float*)d_in[8];
  const float* dmean  = (const float*)d_in[9];
  const float* dstd   = (const float*)d_in[10];
  const float* mW1    = (const float*)d_in[11];
  const float* mb1    = (const float*)d_in[12];
  const float* mW2    = (const float*)d_in[13];
  const float* mb2    = (const float*)d_in[14];
  const float* aW     = (const float*)d_in[15];
  const float* ab     = (const float*)d_in[16];
  const float* xW1    = (const float*)d_in[17];
  const float* xb1    = (const float*)d_in[18];
  const float* xW2    = (const float*)d_in[19];
  const float* uW1    = (const float*)d_in[20];
  const float* ub1    = (const float*)d_in[21];
  const float* uW2    = (const float*)d_in[22];
  const float* ub2    = (const float*)d_in[23];

  // ---- workspace carve (all sub-buffers 16B aligned) ----
  float* hA   = (float*)d_ws;                 // 2048*64 f32
  float* hB   = hA   + NNODE*FEAT;
  float* posA = hB   + NNODE*FEAT;            // 2048*3
  float* posB = posA + NNODE*3;
  h16* hA16  = (h16*)(posB + NNODE*3);        // 2048*64 f16
  h16* hB16  = hA16 + NNODE*FEAT;
  h16* agg16 = hB16 + NNODE*FEAT;             // 2048*64 f16
  h16* wm1   = agg16 + NNODE*FEAT;            // 4*64*192
  h16* wm2   = wm1 + 4*64*KIN;                // 4*64*64
  h16* wx1   = wm2 + 4*64*64;                 // 4*64*64
  h16* wu1   = wx1 + 4*64*64;                 // 4*64*128
  h16* wu2   = wu1 + 4*64*128;                // 4*64*64

  // one-time weight conversion to f16
  egnn_cvt16<<<(4*64*KIN+255)/256, 256, 0, stream>>>(mW1, wm1, 4*64*KIN);
  egnn_cvt16<<<(4*64*64 +255)/256, 256, 0, stream>>>(mW2, wm2, 4*64*64);
  egnn_cvt16<<<(4*64*64 +255)/256, 256, 0, stream>>>(xW1, wx1, 4*64*64);
  egnn_cvt16<<<(4*64*128+255)/256, 256, 0, stream>>>(uW1, wu1, 4*64*128);
  egnn_cvt16<<<(4*64*64 +255)/256, 256, 0, stream>>>(uW2, wu2, 4*64*64);

  const size_t edgeSmem =
      (size_t)(128*LDI + 64*LDI + 128*LDM + 128*LDM + 64*LDM)*sizeof(h16)
    + (size_t)(128*4 + 128*4 + 256)*sizeof(float);

  egnn_embed<<<(NNODE*FEAT+255)/256, 256, 0, stream>>>(t, ptypes, tmean, tstd, hW, hb, hA, hA16);

  float* hbuf [2] = { hA, hB };
  h16*   hbuf16[2] = { hA16, hB16 };
  float* pbuf [2] = { posA, posB };
  int hi = 0;
  const float* pin = x;
  for (int l = 0; l < 4; ++l) {
    float* pout = pbuf[l & 1];
    egnn_edge_wmma<<<NNODE, 256, edgeSmem, stream>>>(
        hbuf16[hi], pin, esrc,
        dmean + l*64, dstd + l*64,
        wm1 + l*64*KIN, mb1 + l*64, wm2 + l*64*64, mb2 + l*64,
        aW + l*64, ab + l, wx1 + l*64*64, xb1 + l*64, xW2 + l*64,
        agg16, pout);
    egnn_node_wmma<<<NNODE/64, 128, 0, stream>>>(
        hbuf[hi], hbuf16[hi], agg16,
        wu1 + l*64*128, ub1 + l*64, wu2 + l*64*64, ub2 + l*64,
        hbuf[1-hi], hbuf16[1-hi]);
    hi ^= 1;
    pin = pout;
  }

  egnn_vel<<<(NNODE*3+255)/256, 256, 0, stream>>>(pin, x, (float*)d_out);
}
